// CrossBatchAttention_20933670601091
// MI455X (gfx1250) — compile-verified, matrix operand order checked by reading, not run on hardware
//
#include <hip/hip_runtime.h>
#include <math.h>

typedef __bf16 bf16;
typedef __attribute__((ext_vector_type(16))) __bf16 v16bf;
typedef __attribute__((ext_vector_type(8)))  float  v8f;
typedef __attribute__((ext_vector_type(4)))  int    v4i;

#define AS1 __attribute__((address_space(1)))
#define AS3 __attribute__((address_space(3)))

constexpr int   Bx    = 4096;
constexpr int   Hx    = 1024;
constexpr int   NHx   = 8;
constexpr int   HDx   = 128;   // Hx / NHx
constexpr int   TOPKx = 32;
constexpr int   GHx   = 256;
constexpr float SCALEx = 0.08838834764831845f; // 1/sqrt(128)
constexpr float NEGBIG = -1.0e30f;

// ---------------------------------------------------------------------------
// CDNA5 async copy: 16B global -> LDS, tracked by ASYNCcnt.
// Builtin prototype (probe-confirmed): (global v4i*, local v4i*, imm, imm)
// ---------------------------------------------------------------------------
__device__ __forceinline__ void async_cp16(const bf16* g, bf16* l) {
#if __has_builtin(__builtin_amdgcn_global_load_async_to_lds_b128)
  __builtin_amdgcn_global_load_async_to_lds_b128((AS1 v4i*)g, (AS3 v4i*)l, 0, 0);
#else
  unsigned lds_off = (unsigned)(size_t)(AS3 bf16*)l;
  asm volatile("global_load_async_to_lds_b128 %0, %1, off"
               :: "v"(lds_off), "v"(g) : "memory");
#endif
}

__device__ __forceinline__ void wait_async0() {
#if __has_builtin(__builtin_amdgcn_s_wait_asynccnt)
  __builtin_amdgcn_s_wait_asynccnt(0);
#else
  asm volatile("s_wait_asynccnt 0x0" ::: "memory");
#endif
}

// ---------------------------------------------------------------------------
// WMMA bf16 fragment load for row-major data, K contiguous.
// 16-bit A/B layout (16x16x32): lane half = lane>>4 selects K sub-block.
// elements 0..7  -> K = k0 + half*8 + e        (contiguous run of 8)
// elements 8..15 -> K = k0 + 16 + half*8 + e-8 (contiguous run of 8)
// ---------------------------------------------------------------------------
__device__ __forceinline__ v16bf frag_ld(const bf16* rowp, int k0, int half) {
  const bf16* p = rowp + k0 + (half << 3);
  v16bf f;
#pragma unroll
  for (int i = 0; i < 8; ++i) { f[i] = p[i]; f[i + 8] = p[16 + i]; }
  return f;
}

__device__ __forceinline__ v8f wmma_bf16(v16bf a, v16bf b, v8f c) {
  return __builtin_amdgcn_wmma_f32_16x16x32_bf16(false, a, false, b,
                                                 (short)0, c, false, false);
}

// ---------------------------------------------------------------------------
// fp32 -> bf16 conversion
// ---------------------------------------------------------------------------
__global__ void k_f32_to_bf16(const float* __restrict__ in, bf16* __restrict__ out, int n) {
  int i = blockIdx.x * 256 + threadIdx.x;
  if (i < n) out[i] = (bf16)in[i];
}

// ---------------------------------------------------------------------------
// Double-buffered 64x64 GEMM body: O = inp(B x H) @ W^T, 2x2 WMMA tiles/wave.
// Per K-chunk (64 rows x 128 k, 16 KB bf16): async-stage chunk k+1 into LDS
// while computing WMMAs on chunk k; s_wait_asynccnt + barrier per iteration.
// block = 128 threads (4 waves); wave (wr,wc) owns 32x32 of the 64x64 tile.
// ---------------------------------------------------------------------------
__device__ __forceinline__ void stage_chunk(const bf16* __restrict__ inp,
                                            bf16* buf, int rowBase, int kc) {
#pragma unroll
  for (int g0 = 0; g0 < 8; ++g0) {
    int g = threadIdx.x + g0 * 128;    // 1024 groups of 8 bf16 (16B)
    int row = g >> 4;                  // 16 groups per 128-wide row
    int c8  = (g & 15) << 3;
    async_cp16(inp + (size_t)(rowBase + row) * Hx + kc * 128 + c8,
               buf + row * 128 + c8);
  }
}

__device__ __forceinline__ void gemm64(const bf16* __restrict__ inp,
                                       const bf16* __restrict__ w,
                                       int rowBase, int colBase0,
                                       v8f acc[2][2]) {
  __shared__ __align__(16) bf16 xs[2][64 * 128];  // 32 KB double buffer
  int lane = threadIdx.x & 31, wave = threadIdx.x >> 5;
  int half = lane >> 4, rc = lane & 15;
  int wr = wave >> 1, wc = wave & 1;

  const bf16* wrow0 = w + (size_t)(colBase0 + wc * 32 + rc) * Hx;
  const bf16* wrow1 = wrow0 + (size_t)16 * Hx;

  stage_chunk(inp, xs[0], rowBase, 0);
  wait_async0();
  __syncthreads();

  for (int kc = 0; kc < Hx / 128; ++kc) {
    int cur = kc & 1;
    if (kc + 1 < Hx / 128) stage_chunk(inp, xs[cur ^ 1], rowBase, kc + 1);

    const bf16* xbuf = xs[cur];
    const bf16* arow0 = xbuf + (wr * 32 + rc) * 128;
    const bf16* arow1 = arow0 + 16 * 128;
#pragma unroll
    for (int ks = 0; ks < 128; ks += 32) {
      v16bf a0 = frag_ld(arow0, ks, half);
      v16bf a1 = frag_ld(arow1, ks, half);
      int kg = kc * 128 + ks;
      v16bf b0 = frag_ld(wrow0, kg, half);
      v16bf b1 = frag_ld(wrow1, kg, half);
      acc[0][0] = wmma_bf16(a0, b0, acc[0][0]);
      acc[0][1] = wmma_bf16(a0, b1, acc[0][1]);
      acc[1][0] = wmma_bf16(a1, b0, acc[1][0]);
      acc[1][1] = wmma_bf16(a1, b1, acc[1][1]);
    }
    wait_async0();      // chunk kc+1 landed; safe to flip buffers
    __syncthreads();
  }
}

// ---------------------------------------------------------------------------
// QKV projection.  grid: (Bx/64, Hx/64, 3)  block: 128
// Output layout: [3][NH][B][HD] bf16 (per-head, K-major rows for attention)
// ---------------------------------------------------------------------------
__global__ void k_qkv_gemm(const bf16* __restrict__ xb,
                           const bf16* __restrict__ wq,
                           const bf16* __restrict__ wk,
                           const bf16* __restrict__ wv,
                           bf16* __restrict__ qkv) {
  const bf16* w = (blockIdx.z == 0) ? wq : (blockIdx.z == 1) ? wk : wv;
  bf16* outz = qkv + (size_t)blockIdx.z * NHx * Bx * HDx;
  int rowBase = blockIdx.x * 64;
  int colBase0 = blockIdx.y * 64;

  v8f acc[2][2] = {};
  gemm64(xb, w, rowBase, colBase0, acc);

  int lane = threadIdx.x & 31, wave = threadIdx.x >> 5;
  int half = lane >> 4, rc = lane & 15;
  int wr = wave >> 1, wc = wave & 1;
#pragma unroll
  for (int i = 0; i < 2; ++i)
#pragma unroll
    for (int j = 0; j < 2; ++j) {
      int rBase = rowBase + wr * 32 + i * 16;
      int col = colBase0 + wc * 32 + j * 16 + rc;  // 16-wide tile: one head
      int h = col / HDx, d = col % HDx;
#pragma unroll
      for (int r = 0; r < 8; ++r)
        outz[((size_t)h * Bx + rBase + r + half * 8) * HDx + d] = (bf16)acc[i][j][r];
    }
}

// ---------------------------------------------------------------------------
// Output projection: cbp = attn_out @ Wo^T (fp32).  grid: (Bx/64, Hx/64)
// ---------------------------------------------------------------------------
__global__ void k_out_proj(const bf16* __restrict__ ab, const bf16* __restrict__ wo,
                           float* __restrict__ cbp) {
  int rowBase = blockIdx.x * 64;
  int colBase0 = blockIdx.y * 64;

  v8f acc[2][2] = {};
  gemm64(ab, wo, rowBase, colBase0, acc);

  int lane = threadIdx.x & 31, wave = threadIdx.x >> 5;
  int half = lane >> 4, rc = lane & 15;
  int wr = wave >> 1, wc = wave & 1;
#pragma unroll
  for (int i = 0; i < 2; ++i)
#pragma unroll
    for (int j = 0; j < 2; ++j) {
      int rBase = rowBase + wr * 32 + i * 16;
      int col = colBase0 + wc * 32 + j * 16 + rc;
#pragma unroll
      for (int r = 0; r < 8; ++r)
        cbp[(size_t)(rBase + r + half * 8) * Hx + col] = acc[i][j][r];
    }
}

// ---------------------------------------------------------------------------
// Attention: one workgroup = (head, 16 query rows).
//   Phase 1: Q*K^T via WMMA -> 16x4096 fp32 score stripe in LDS (256 KB)
//   Phase 2: per-row top-32 by iterated wave-argmax extraction
//   Phase 3: softmax over the 32 survivors + sparse V gather
// grid: (Bx/16, NH)  block: 256 (8 waves)  dynamic LDS: 16*Bx*4 bytes
// ---------------------------------------------------------------------------
__global__ void k_attention(const bf16* __restrict__ qb,
                            const bf16* __restrict__ kb,
                            const bf16* __restrict__ vb,
                            const unsigned char* __restrict__ amask,
                            bf16* __restrict__ attn_out /* [B][H] bf16 */) {
  extern __shared__ float S[];                 // 16 x 4096 fp32 scores
  __shared__ __align__(16) bf16 qs[16 * HDx];  // 4 KB q tile
  __shared__ float topv[16][TOPKx];
  __shared__ int   topi[16][TOPKx];

  int h = blockIdx.y;
  int qbase = blockIdx.x * 16;

  // async-stage the q tile (16 x 128 bf16 = 4 KB)
  {
    int g = threadIdx.x;                       // 256 groups of 8 bf16
    int row = g >> 4, c8 = (g & 15) << 3;
    async_cp16(qb + ((size_t)h * Bx + qbase + row) * HDx + c8, qs + row * HDx + c8);
    wait_async0();
  }
  __syncthreads();

  int lane = threadIdx.x & 31, wave = threadIdx.x >> 5;
  int half = lane >> 4, rc = lane & 15;

  // A fragments (q tile) loaded once, reused across all key tiles of the wave
  v16bf aq[4];
#pragma unroll
  for (int kk = 0; kk < 4; ++kk) aq[kk] = frag_ld(qs + rc * HDx, kk * 32, half);

  for (int kt = wave; kt < Bx / 16; kt += 8) {
    int gk = kt * 16 + rc;
    const bf16* krow = kb + ((size_t)h * Bx + gk) * HDx;
    v8f acc = {};
#pragma unroll
    for (int kk = 0; kk < 4; ++kk) {
      v16bf bk = frag_ld(krow, kk * 32, half);
      acc = wmma_bf16(aq[kk], bk, acc);
    }
    bool valid = amask[gk] != 0;
#pragma unroll
    for (int r = 0; r < 8; ++r) {
      int m = r + half * 8;
      float s = acc[r] * SCALEx;
      if (!valid || (qbase + m) == gk) s = NEGBIG;  // key mask + diagonal mask
      S[m * Bx + gk] = s;
    }
  }
  __syncthreads();

  // each wave owns rows {2*wave, 2*wave+1}
  for (int rr = 0; rr < 2; ++rr) {
    int m = wave * 2 + rr;
    float* row = S + m * Bx;

    for (int round = 0; round < TOPKx; ++round) {
      float best = -3.0e38f; int bidx = 0;
      for (int i = 0; i < Bx / 32; ++i) {
        int c = lane + (i << 5);
        float v = row[c];
        if (v > best) { best = v; bidx = c; }
      }
#pragma unroll
      for (int off = 16; off > 0; off >>= 1) {
        float ov = __shfl_xor(best, off);
        int   oi = __shfl_xor(bidx, off);
        if (ov > best) { best = ov; bidx = oi; }
      }
      if (lane == 0) {
        topv[m][round] = best;
        topi[m][round] = bidx;
        row[bidx] = NEGBIG;   // extracted => removed from further rounds
      }
    }

    // softmax over exactly the 32 survivors (== threshold semantics)
    float mval = topv[m][0];                   // extracted in descending order
    float e = __expf(topv[m][lane] - mval);
    float ssum = e;
#pragma unroll
    for (int off = 16; off > 0; off >>= 1) ssum += __shfl_xor(ssum, off);
    topv[m][lane] = e / ssum;                  // stash probability

    // sparse P*V: 32 weighted V rows, lanes parallel over d (4 dims each)
    float accd[4] = {0.f, 0.f, 0.f, 0.f};
    for (int l = 0; l < TOPKx; ++l) {
      float pl = topv[m][l];
      int   ki = topi[m][l];
      const bf16* vrow = vb + ((size_t)h * Bx + ki) * HDx;
#pragma unroll
      for (int j = 0; j < 4; ++j)
        accd[j] += pl * (float)vrow[lane + (j << 5)];
    }
#pragma unroll
    for (int j = 0; j < 4; ++j)
      attn_out[(size_t)(qbase + m) * Hx + h * HDx + lane + (j << 5)] = (bf16)accd[j];
  }
}

// ---------------------------------------------------------------------------
// Dual LayerNorm: ln_h = LN(x), ln_a = LN(cbp).  grid: Bx, block: 256
// ---------------------------------------------------------------------------
__global__ void k_layernorms(const float* __restrict__ x, const float* __restrict__ cbp,
                             const float* __restrict__ lhw, const float* __restrict__ lhb,
                             const float* __restrict__ law, const float* __restrict__ lab,
                             float* __restrict__ lnh, float* __restrict__ lna) {
  __shared__ float sred[8], qred[8], mm[2];
  int b = blockIdx.x;
  int lane = threadIdx.x & 31, wave = threadIdx.x >> 5;

  const float* rows[2] = { x + (size_t)b * Hx, cbp + (size_t)b * Hx };
  float*       outs[2] = { lnh + (size_t)b * Hx, lna + (size_t)b * Hx };
  const float* gw[2]   = { lhw, law };
  const float* gb[2]   = { lhb, lab };

  for (int t = 0; t < 2; ++t) {
    float s = 0.f, sq = 0.f;
    for (int j = threadIdx.x; j < Hx; j += blockDim.x) {
      float v = rows[t][j]; s += v; sq += v * v;
    }
#pragma unroll
    for (int off = 16; off > 0; off >>= 1) {
      s += __shfl_xor(s, off); sq += __shfl_xor(sq, off);
    }
    if (lane == 0) { sred[wave] = s; qred[wave] = sq; }
    __syncthreads();
    if (threadIdx.x == 0) {
      float ts = 0.f, tq = 0.f;
      for (int w2 = 0; w2 < 8; ++w2) { ts += sred[w2]; tq += qred[w2]; }
      float mean = ts / Hx;
      mm[0] = mean;
      mm[1] = tq / Hx - mean * mean;
    }
    __syncthreads();
    float mean = mm[0];
    float inv  = rsqrtf(mm[1] + 1e-5f);
    for (int j = threadIdx.x; j < Hx; j += blockDim.x)
      outs[t][j] = (rows[t][j] - mean) * inv * gw[t][j] + gb[t][j];
    __syncthreads();
  }
}

// ---------------------------------------------------------------------------
// Gate stage 1: hgate = gelu([ln_h | ln_a | ln_h*ln_a] @ g1_w^T + g1_b)
// grid: Bx, block: 256 (8 waves, 32 outputs each, wave-dot of length 3072)
// ---------------------------------------------------------------------------
__global__ void k_gate1(const float* __restrict__ lnh, const float* __restrict__ lna,
                        const float* __restrict__ g1w, const float* __restrict__ g1b,
                        float* __restrict__ hg) {
  __shared__ float lh[Hx], la[Hx];
  int b = blockIdx.x;
  for (int j = threadIdx.x; j < Hx; j += blockDim.x) {
    lh[j] = lnh[(size_t)b * Hx + j];
    la[j] = lna[(size_t)b * Hx + j];
  }
  __syncthreads();

  int lane = threadIdx.x & 31, wave = threadIdx.x >> 5;
  for (int gi = 0; gi < GHx / 8; ++gi) {
    int g = wave * (GHx / 8) + gi;
    const float* wrow = g1w + (size_t)g * (3 * Hx);
    float acc = 0.f;
    for (int j = lane; j < Hx; j += 32) {
      float a = lh[j], c = la[j];
      acc += a * wrow[j] + c * wrow[Hx + j] + a * c * wrow[2 * Hx + j];
    }
#pragma unroll
    for (int off = 16; off > 0; off >>= 1) acc += __shfl_xor(acc, off);
    if (lane == 0) {
      float hv = acc + g1b[g];
      hg[(size_t)b * GHx + g] = 0.5f * hv * (1.f + erff(hv * 0.70710678118654752f));
    }
  }
}

// ---------------------------------------------------------------------------
// Gate stage 2 + residual: out = x + sigmoid(hgate @ g2_w^T + g2_b) * cbp
// grid: Bx, block: 256
// ---------------------------------------------------------------------------
__global__ void k_gate2_out(const float* __restrict__ x, const float* __restrict__ cbp,
                            const float* __restrict__ hg, const float* __restrict__ g2w,
                            const float* __restrict__ g2b, float* __restrict__ out) {
  __shared__ float hgs[GHx];
  int b = blockIdx.x;
  for (int j = threadIdx.x; j < GHx; j += blockDim.x) hgs[j] = hg[(size_t)b * GHx + j];
  __syncthreads();

  for (int o = threadIdx.x; o < Hx; o += blockDim.x) {
    const float* wrow = g2w + (size_t)o * GHx;
    float acc = 0.f;
#pragma unroll 8
    for (int j = 0; j < GHx; ++j) acc += hgs[j] * wrow[j];
    float gate = 1.f / (1.f + __expf(-(acc + g2b[o])));
    out[(size_t)b * Hx + o] = x[(size_t)b * Hx + o] + gate * cbp[(size_t)b * Hx + o];
  }
}

// ---------------------------------------------------------------------------
extern "C" void kernel_launch(void* const* d_in, const int* in_sizes, int n_in,
                              void* d_out, int out_size, void* d_ws, size_t ws_size,
                              hipStream_t stream) {
  (void)in_sizes; (void)n_in; (void)out_size; (void)ws_size;

  const float* x            = (const float*)d_in[0];
  const unsigned char* mask = (const unsigned char*)d_in[1];
  const float* Wq  = (const float*)d_in[2];
  const float* Wk  = (const float*)d_in[3];
  const float* Wv  = (const float*)d_in[4];
  const float* Wo  = (const float*)d_in[5];
  const float* lhw = (const float*)d_in[6];
  const float* lhb = (const float*)d_in[7];
  const float* law = (const float*)d_in[8];
  const float* lab = (const float*)d_in[9];
  const float* g1w = (const float*)d_in[10];
  const float* g1b = (const float*)d_in[11];
  const float* g2w = (const float*)d_in[12];
  const float* g2b = (const float*)d_in[13];
  float* out = (float*)d_out;

  // workspace carve-out (256B aligned)
  char* ws = (char*)d_ws;
  size_t off = 0;
  auto carve = [&](size_t bytes) {
    void* p = ws + off;
    off += (bytes + 255) & ~(size_t)255;
    return p;
  };
  bf16*  xb   = (bf16*)carve((size_t)Bx * Hx * 2);
  bf16*  wqb  = (bf16*)carve((size_t)Hx * Hx * 2);
  bf16*  wkb  = (bf16*)carve((size_t)Hx * Hx * 2);
  bf16*  wvb  = (bf16*)carve((size_t)Hx * Hx * 2);
  bf16*  wob  = (bf16*)carve((size_t)Hx * Hx * 2);
  bf16*  qkvb = (bf16*)carve((size_t)3 * NHx * Bx * HDx * 2);  // [3][NH][B][HD]
  bf16*  ab   = (bf16*)carve((size_t)Bx * Hx * 2);             // attn out bf16
  float* cbp  = (float*)carve((size_t)Bx * Hx * 4);            // post-Wo fp32
  float* lnh  = (float*)carve((size_t)Bx * Hx * 4);
  float* lna  = (float*)carve((size_t)Bx * Hx * 4);
  float* hg   = (float*)carve((size_t)Bx * GHx * 4);

  // 1) bf16 conversions
  int nxh = Bx * Hx, nww = Hx * Hx;
  k_f32_to_bf16<<<(nxh + 255) / 256, 256, 0, stream>>>(x,  xb,  nxh);
  k_f32_to_bf16<<<(nww + 255) / 256, 256, 0, stream>>>(Wq, wqb, nww);
  k_f32_to_bf16<<<(nww + 255) / 256, 256, 0, stream>>>(Wk, wkb, nww);
  k_f32_to_bf16<<<(nww + 255) / 256, 256, 0, stream>>>(Wv, wvb, nww);
  k_f32_to_bf16<<<(nww + 255) / 256, 256, 0, stream>>>(Wo, wob, nww);

  // 2) fused QKV projection (async double-buffered WMMA GEMM, 2x2 tiles/wave)
  k_qkv_gemm<<<dim3(Bx / 64, Hx / 64, 3), 128, 0, stream>>>(xb, wqb, wkb, wvb, qkvb);

  // 3) attention with 256KB LDS score stripe (WMMA scores + sparse top-32 PV)
  const bf16* qB = qkvb;
  const bf16* kB = qkvb + (size_t)NHx * Bx * HDx;
  const bf16* vB = qkvb + (size_t)2 * NHx * Bx * HDx;
  size_t smem = (size_t)16 * Bx * sizeof(float);   // 256 KB dynamic
  (void)hipFuncSetAttribute(reinterpret_cast<const void*>(k_attention),
                            hipFuncAttributeMaxDynamicSharedMemorySize, (int)smem);
  k_attention<<<dim3(Bx / 16, NHx), 256, smem, stream>>>(qB, kB, vB, mask, ab);

  // 4) output projection (async double-buffered WMMA GEMM)
  k_out_proj<<<dim3(Bx / 64, Hx / 64), 128, 0, stream>>>(ab, wob, cbp);

  // 5) layer norms
  k_layernorms<<<Bx, 256, 0, stream>>>(x, cbp, lhw, lhb, law, lab, lnh, lna);

  // 6) gate MLP + gated residual
  k_gate1<<<Bx, 256, 0, stream>>>(lnh, lna, g1w, g1b, hg);
  k_gate2_out<<<Bx, 256, 0, stream>>>(x, cbp, hg, g2w, g2b, out);
}
